// QwenExtension_11768210391391
// MI455X (gfx1250) — matrix operand order repaired and checked by reading, not compiled, not run on hardware
//
#include <hip/hip_runtime.h>

typedef __bf16 bf16_t;
typedef __attribute__((ext_vector_type(16))) __bf16 bf16x16;
typedef __attribute__((ext_vector_type(8)))  __bf16 bf16x8;
typedef __attribute__((ext_vector_type(4)))  __bf16 bf16x4;
typedef __attribute__((ext_vector_type(8)))  float  f32x8;

#define L_ 28
#define E_ 1024
#define H_ 8
#define D_ 128
#define B_ 2
#define S_ 1024
#define NROWS 2048          // B_*S_
#define EPS_ 1e-6f
#define THETA_LOG 13.815510557964274f   // ln(1e6)

#define BM 128
#define BK 64
#define LDA 72              // bf16 elems per tile row (144 B, 16B-aligned, pads banks)
#define LDK 136             // bf16 elems per epilogue row (272 B, 16B-aligned)

// ---------------------------------------------------------------------------
// Kernel 1: per-row mean of x^2 over E (one wave32 per row)
// ---------------------------------------------------------------------------
__global__ __launch_bounds__(256) void rowstats_kernel(
    const float* __restrict__ x, float* __restrict__ mean_out) {
  const int wave = threadIdx.x >> 5;
  const int lane = threadIdx.x & 31;
  const int row  = blockIdx.x * 8 + wave;          // grid 256 -> 2048 rows
  const float4* xr = (const float4*)(x + (size_t)row * E_);
  float s = 0.f;
#pragma unroll
  for (int i = 0; i < 8; ++i) {
    float4 v = xr[i * 32 + lane];
    s += v.x * v.x + v.y * v.y + v.z * v.z + v.w * v.w;
  }
#pragma unroll
  for (int off = 16; off > 0; off >>= 1) s += __shfl_xor(s, off, 32);
  if (lane == 0) mean_out[row] = s * (1.0f / E_);
}

// ---------------------------------------------------------------------------
// Kernel 2: fused RMSnorm -> bf16 WMMA GEMM -> (k: head-RMS + RoPE | v: copy)
// grid = (16 row tiles, 8 heads, 28 layers * 2 {k,v}); block = 256 (8 waves)
// Software-pipelined: stage i+1's global tiles are fetched into VGPRs while
// the WMMAs for stage i run out of LDS.
// ---------------------------------------------------------------------------
__global__ __launch_bounds__(256) void kv_gemm_kernel(
    const float* __restrict__ x,        // (2048,1024)
    const float* __restrict__ row_mean, // (2048)
    const float* __restrict__ scales,   // (28)
    const float* __restrict__ ln_w,     // (28,1024)
    const float* __restrict__ Wk,       // (28,1024,1024)
    const float* __restrict__ Wv,       // (28,1024,1024)
    const float* __restrict__ kn_w,     // (28,128)
    float* __restrict__ out)            // (2,28,2,8,1024,128)
{
  __shared__ union {
    struct {
      bf16_t A[BM * LDA];
      bf16_t W[BM * LDA];
    } g;
    struct {
      bf16_t K[BM * LDK];
      float  rinv[BM];
    } e;
  } sm;

  const int bx  = blockIdx.x;           // 0..15 row tile
  const int h   = blockIdx.y;           // 0..7 head
  const int l   = blockIdx.z >> 1;      // layer
  const int kv  = blockIdx.z & 1;       // 0 = K proj, 1 = V proj
  const int tid = threadIdx.x;

  const int r0 = bx * BM;
  const float* Wbase = (kv ? Wv : Wk) + ((size_t)l * E_ + h * D_) * (size_t)E_;

  f32x8 acc[2][4];
#pragma unroll
  for (int i = 0; i < 2; ++i)
#pragma unroll
    for (int j = 0; j < 4; ++j) { f32x8 z = {}; acc[i][j] = z; }

  const int wave = tid >> 5;
  const int lane = tid & 31;
  const int m0w  = (wave & 3) * 32;     // wave row base within 128
  const int n0w  = (wave >> 2) * 64;    // wave col base within 128
  const int half = lane >> 4;
  const int l16  = lane & 15;

  const int ldrow = tid >> 4;           // 16 threads per row, 16 rows per pass
  const int ldcol = (tid & 15) * 4;     // float column within 64-wide chunk

  // per-row combined scale for the 8 rows this thread stages (loop-invariant):
  // f = s * rsqrt(s^2 * mean(x^2) + eps)
  float fReg[8];
  {
    const float sc = scales[l];
#pragma unroll
    for (int p = 0; p < 8; ++p) {
      const float m = row_mean[r0 + p * 16 + ldrow];
      fReg[p] = sc * rsqrtf(sc * sc * m + EPS_);
    }
  }

  // ---- software pipeline: preload stage 0 into registers -------------------
  float4 aReg[8], wReg[8], lwReg;
  lwReg = *(const float4*)(ln_w + (size_t)l * E_ + ldcol);
#pragma unroll
  for (int p = 0; p < 8; ++p) {
    const int r = p * 16 + ldrow;
    aReg[p] = *(const float4*)(x + (size_t)(r0 + r) * E_ + ldcol);
    wReg[p] = *(const float4*)(Wbase + (size_t)r * E_ + ldcol);
  }

  for (int kc = 0; kc < E_; kc += BK) {
    // convert current stage registers -> LDS (bf16, ln_w * rowscale applied)
#pragma unroll
    for (int p = 0; p < 8; ++p) {
      const int r  = p * 16 + ldrow;
      const float fr = fReg[p];
      const float4 a  = aReg[p];
      const float4 w4 = wReg[p];
      bf16x4 av = { (bf16_t)(a.x * lwReg.x * fr), (bf16_t)(a.y * lwReg.y * fr),
                    (bf16_t)(a.z * lwReg.z * fr), (bf16_t)(a.w * lwReg.w * fr) };
      *(bf16x4*)&sm.g.A[r * LDA + ldcol] = av;
      bf16x4 wv = { (bf16_t)w4.x, (bf16_t)w4.y, (bf16_t)w4.z, (bf16_t)w4.w };
      *(bf16x4*)&sm.g.W[r * LDA + ldcol] = wv;
    }
    __syncthreads();

    // prefetch next stage (overlaps with the WMMAs below)
    if (kc + BK < E_) {
      const int kn = kc + BK;
      lwReg = *(const float4*)(ln_w + (size_t)l * E_ + kn + ldcol);
#pragma unroll
      for (int p = 0; p < 8; ++p) {
        const int r = p * 16 + ldrow;
        aReg[p] = *(const float4*)(x + (size_t)(r0 + r) * E_ + kn + ldcol);
        wReg[p] = *(const float4*)(Wbase + (size_t)r * E_ + kn + ldcol);
      }
    }

#pragma unroll
    for (int kk = 0; kk < BK; kk += 32) {
      // A fragments (ISA 16-bit A layout: e0-7 <- K+8*half, e8-15 <- K+16+8*half)
      bf16x16 afrag[2];
#pragma unroll
      for (int mt = 0; mt < 2; ++mt) {
        const bf16_t* base = &sm.g.A[(m0w + mt * 16 + l16) * LDA + kk + 8 * half];
        bf16x8 lo = *(const bf16x8*)base;
        bf16x8 hi = *(const bf16x8*)(base + 16);
        afrag[mt] = __builtin_shufflevector(lo, hi, 0, 1, 2, 3, 4, 5, 6, 7,
                                            8, 9, 10, 11, 12, 13, 14, 15);
      }
      // All four B fragments up front (distinct regs -> WMMAs issue back-to-back)
      bf16x16 bfrag[4];
#pragma unroll
      for (int nt = 0; nt < 4; ++nt) {
        const bf16_t* bbase =
            &sm.g.W[(n0w + nt * 16 + l16) * LDA + kk + 8 * half];
        bf16x8 blo = *(const bf16x8*)bbase;
        bf16x8 bhi = *(const bf16x8*)(bbase + 16);
        bfrag[nt] = __builtin_shufflevector(blo, bhi, 0, 1, 2, 3, 4, 5, 6, 7,
                                            8, 9, 10, 11, 12, 13, 14, 15);
      }
#pragma unroll
      for (int nt = 0; nt < 4; ++nt)
#pragma unroll
        for (int mt = 0; mt < 2; ++mt)
          acc[mt][nt] = __builtin_amdgcn_wmma_f32_16x16x32_bf16(
              false, afrag[mt], false, bfrag[nt], (short)0, acc[mt][nt],
              false, false);
    }
    __syncthreads();
  }

  // -------------------------- epilogue ------------------------------------
  if (kv == 1) {
    // V: straight store. C layout: VGPR j -> row j+8*half, col = lane%16.
#pragma unroll
    for (int mt = 0; mt < 2; ++mt)
#pragma unroll
      for (int nt = 0; nt < 4; ++nt)
#pragma unroll
        for (int j = 0; j < 8; ++j) {
          const int lr = m0w + mt * 16 + half * 8 + j;
          const int d  = n0w + nt * 16 + l16;
          const int gr = r0 + lr;
          const int b = gr >> 10, s = gr & 1023;
          const size_t off =
              ((((size_t)(L_ + l) * B_ + b) * H_ + h) * S_ + s) * D_ + d;
          __builtin_nontemporal_store(acc[mt][nt][j], out + off);
        }
    return;
  }

  // K: stash raw projection in LDS (bf16) for cross-wave RMS + RoPE pairing
#pragma unroll
  for (int mt = 0; mt < 2; ++mt)
#pragma unroll
    for (int nt = 0; nt < 4; ++nt)
#pragma unroll
      for (int j = 0; j < 8; ++j) {
        const int lr = m0w + mt * 16 + half * 8 + j;
        const int d  = n0w + nt * 16 + l16;
        sm.e.K[lr * LDK + d] = (bf16_t)acc[mt][nt][j];
      }
  __syncthreads();

  if (tid < BM) {
    const bf16x8* rowp = (const bf16x8*)&sm.e.K[tid * LDK];
    float ss = 0.f;
#pragma unroll
    for (int i = 0; i < 16; ++i) {
      bf16x8 v = rowp[i];
#pragma unroll
      for (int j = 0; j < 8; ++j) { float fv = (float)v[j]; ss += fv * fv; }
    }
    sm.e.rinv[tid] = rsqrtf(ss * (1.0f / D_) + EPS_);
  }
  __syncthreads();

  // head-RMS * kn_w, then RoPE: out[d] = k[d]*cos + (d<64 ? -k[d+64] : k[d-64])*sin
  const int   d    = tid & 127;                 // fixed per thread -> coalesced
  const int   fi   = d & 63;
  const float knd  = kn_w[l * D_ + d];
  const float knp  = kn_w[l * D_ + (d ^ 64)];
  const float sign = (d < 64) ? -1.0f : 1.0f;
  const float invf = __expf((float)fi * (-THETA_LOG / 64.0f));
#pragma unroll 4
  for (int i = 0; i < 64; ++i) {
    const int lr = (tid >> 7) + 2 * i;
    float kd = (float)sm.e.K[lr * LDK + d];
    float kp = (float)sm.e.K[lr * LDK + (d ^ 64)];
    const float inv = sm.e.rinv[lr];
    kd *= inv * knd;
    kp *= inv * knp;
    const int gr = r0 + lr;
    const int b = gr >> 10, s = gr & 1023;
    float sn, cs;
    __sincosf((float)s * invf, &sn, &cs);
    const size_t off = ((((size_t)l * B_ + b) * H_ + h) * S_ + s) * D_ + d;
    __builtin_nontemporal_store(kd * cs + sign * kp * sn, out + off);
  }
}

// ---------------------------------------------------------------------------
extern "C" void kernel_launch(void* const* d_in, const int* in_sizes, int n_in,
                              void* d_out, int out_size, void* d_ws, size_t ws_size,
                              hipStream_t stream) {
  const float* x      = (const float*)d_in[0];   // image_context (2,1024,1024)
  const float* scales = (const float*)d_in[1];   // (28)
  const float* ln_w   = (const float*)d_in[2];   // (28,1024)
  const float* Wk     = (const float*)d_in[3];   // (28,1024,1024)
  const float* Wv     = (const float*)d_in[4];   // (28,1024,1024)
  const float* kn_w   = (const float*)d_in[5];   // (28,128)
  float* out      = (float*)d_out;
  float* row_mean = (float*)d_ws;                // 2048 floats of scratch

  rowstats_kernel<<<256, 256, 0, stream>>>(x, row_mean);

  dim3 grid(NROWS / BM, H_, L_ * 2);             // (16, 8, 56)
  kv_gemm_kernel<<<grid, 256, 0, stream>>>(x, row_mean, scales, ln_w,
                                           Wk, Wv, kn_w, out);
}
